// MultiHeadedAttention_12816182411858
// MI455X (gfx1250) — matrix operand order repaired and checked by reading, not compiled, not run on hardware
//
#include <hip/hip_runtime.h>
#include <hip/hip_bf16.h>
#include <math.h>

// ---------------------------------------------------------------------------
// MultiHeadedAttention forward for MI455X (gfx1250, wave32, WMMA)
// B=4, S=2048, D=1024, H=16, DK=64
// Pipeline: f32->f16 convert + mask bitpack -> QKV GEMMs (wmma, double-
// buffered async-LDS) -> flash attention (wmma, packed mask, double-buffered
// chunks) -> output GEMM (f32 out)
// ---------------------------------------------------------------------------

#define BB  4
#define SS  2048
#define DD  1024
#define HH  16
#define DKK 64

// gfx1250 async global->LDS DMA path (ASYNCcnt-tracked).
#define ASYNC_STAGE 1

typedef __attribute__((ext_vector_type(16))) _Float16 v16h;
typedef __attribute__((ext_vector_type(8)))  _Float16 v8h;
typedef __attribute__((ext_vector_type(8)))  float    v8f;

// 16-byte global->LDS copy: async DMA if enabled, else through VGPRs.
__device__ __forceinline__ void stage16(const _Float16* gsrc, _Float16* ldst) {
#if ASYNC_STAGE
  const unsigned laddr = (unsigned)(unsigned long long)(const void*)ldst;
  asm volatile("global_load_async_to_lds_b128 %0, %1, off"
               :: "v"(laddr), "v"((unsigned long long)gsrc)
               : "memory");
#else
  *(v8h*)ldst = *(const v8h*)gsrc;
#endif
}

__device__ __forceinline__ void stage_wait() {
#if ASYNC_STAGE
  asm volatile("s_wait_asynccnt 0x0" ::: "memory");
#endif
}

// ---------------------------------------------------------------------------
// fp32 -> fp16 conversion (grid-stride)
// ---------------------------------------------------------------------------
__global__ void f32_to_f16_kernel(const float* __restrict__ in,
                                  _Float16* __restrict__ out, size_t n) {
  size_t i = (size_t)blockIdx.x * blockDim.x + threadIdx.x;
  size_t stride = (size_t)gridDim.x * blockDim.x;
  for (; i < n; i += stride) out[i] = (_Float16)in[i];
}

// ---------------------------------------------------------------------------
// Bit-pack bool mask: 32 bytes -> one u32 via wave32 ballot.
// ---------------------------------------------------------------------------
__global__ void mask_pack_kernel(const unsigned char* __restrict__ mask,
                                 unsigned* __restrict__ mpack, size_t nbytes) {
  size_t i = (size_t)blockIdx.x * blockDim.x + threadIdx.x;
  const size_t stride = (size_t)gridDim.x * blockDim.x;  // multiple of 32
  for (; i < nbytes; i += stride) {
    const unsigned long long b = __ballot(mask[i] != 0);
    if ((threadIdx.x & 31) == 0) mpack[i >> 5] = (unsigned)b;
  }
}

// ---------------------------------------------------------------------------
// C[m,n] = sum_k A[m,k] * W[n,k]   (x @ W^T, torch Linear convention)
// Block tile 256x128, k-step 32; 8 waves x 64x64 sub-tiles.
// Double-buffered LDS: async DMA of step n+1 overlaps the 16 WMMAs of step n.
// ---------------------------------------------------------------------------
template <typename OutT>
__global__ __launch_bounds__(256) void gemm_xwT_kernel(
    const _Float16* __restrict__ A, const _Float16* __restrict__ W,
    OutT* __restrict__ C, int M, int N, int Kd) {
  __shared__ __align__(16) _Float16 As[2][256 * 40];  // stride 40 halves
  __shared__ __align__(16) _Float16 Ws[2][128 * 40];

  const int t = threadIdx.x;
  const int wave = t >> 5, lane = t & 31;
  const int lh = lane & 15, half = lane >> 4;
  const int wm = (wave & 3) * 64;   // wave M offset inside block tile
  const int wn = (wave >> 2) * 64;  // wave N offset inside block tile
  const int m0 = blockIdx.y * 256, n0 = blockIdx.x * 128;

  auto stage_tiles = [&](int k0, int buf) {
    for (int c = t; c < 1024; c += 256) {  // A: 256 rows x 32 halves
      const int row = c >> 2, c8 = (c & 3) * 8;
      stage16(&A[(size_t)(m0 + row) * Kd + k0 + c8], &As[buf][row * 40 + c8]);
    }
    for (int c = t; c < 512; c += 256) {   // W: 128 rows x 32 halves
      const int row = c >> 2, c8 = (c & 3) * 8;
      stage16(&W[(size_t)(n0 + row) * Kd + k0 + c8], &Ws[buf][row * 40 + c8]);
    }
  };

  v8f acc[4][4];
#pragma unroll
  for (int mt = 0; mt < 4; ++mt)
#pragma unroll
    for (int nt = 0; nt < 4; ++nt)
#pragma unroll
      for (int i = 0; i < 8; ++i) acc[mt][nt][i] = 0.f;

  stage_tiles(0, 0);  // prologue

  for (int k0 = 0; k0 < Kd; k0 += 32) {
    const int cur = (k0 >> 5) & 1;
    stage_wait();      // this step's tiles resident in LDS
    __syncthreads();   // ... for all waves; and no wave still reads buf cur^1
    if (k0 + 32 < Kd) stage_tiles(k0 + 32, cur ^ 1);  // DMA next step
    if (k0 + 64 < Kd) {  // far prefetch two steps out (L2 warm)
      __builtin_prefetch(&A[(size_t)(m0 + t) * Kd + k0 + 64], 0, 0);
      __builtin_prefetch(&W[(size_t)(n0 + (t >> 1)) * Kd + k0 + 64], 0, 0);
    }

    // A fragments: lane holds row M=lh; K 0..7/16..23 (half=0) or +8
    v16h af[4];
#pragma unroll
    for (int mt = 0; mt < 4; ++mt) {
      const _Float16* p = &As[cur][(wm + mt * 16 + lh) * 40 + half * 8];
      v8h lo = *(const v8h*)p;
      v8h hi = *(const v8h*)(p + 16);
#pragma unroll
      for (int i = 0; i < 8; ++i) { af[mt][i] = lo[i]; af[mt][8 + i] = hi[i]; }
    }
    // B fragments: lane holds col N=lh; K 0..7/16..23 (half=0) or +8
#pragma unroll
    for (int nt = 0; nt < 4; ++nt) {
      const _Float16* p = &Ws[cur][(wn + nt * 16 + lh) * 40 + half * 8];
      v8h lo = *(const v8h*)p;
      v8h hi = *(const v8h*)(p + 16);
      v16h bf;
#pragma unroll
      for (int i = 0; i < 8; ++i) { bf[i] = lo[i]; bf[8 + i] = hi[i]; }
#pragma unroll
      for (int mt = 0; mt < 4; ++mt)
        acc[mt][nt] = __builtin_amdgcn_wmma_f32_16x16x32_f16(
            false, af[mt], false, bf, (short)0, acc[mt][nt], false, false);
    }
  }

  // C layout: vgpr r, lane L -> (M = r + 8*(L>>4), N = L&15)
#pragma unroll
  for (int mt = 0; mt < 4; ++mt)
#pragma unroll
    for (int nt = 0; nt < 4; ++nt)
#pragma unroll
      for (int r = 0; r < 8; ++r) {
        const size_t row = (size_t)(m0 + wm + mt * 16 + r + 8 * half);
        C[row * N + n0 + wn + nt * 16 + lh] = (OutT)acc[mt][nt][r];
      }
}

// ---------------------------------------------------------------------------
// Flash attention: one block per (b, h, 64-query tile); 4 waves x 16 rows.
// Double-buffered 32-key chunks: K via async DMA, V transposed through VGPRs;
// packed-bit mask; online softmax; fp32 WMMA accumulation.
// ---------------------------------------------------------------------------
__global__ __launch_bounds__(128) void flash_attn_kernel(
    const _Float16* __restrict__ Q, const _Float16* __restrict__ K,
    const _Float16* __restrict__ V, const unsigned* __restrict__ Mpk,
    _Float16* __restrict__ O) {
  __shared__ __align__(16) _Float16 Ks[2][32 * 72];   // K chunk [key][dk]
  __shared__ __align__(16) _Float16 Vt[2][DKK * 40];  // V chunk transposed
  __shared__ __align__(16) _Float16 Pb[4 * 16 * 40];  // per-wave P tile

  const int t = threadIdx.x;
  const int wave = t >> 5, lane = t & 31;
  const int lh = lane & 15, half = lane >> 4;
  const int b = blockIdx.z, h = blockIdx.y;
  const int q0 = blockIdx.x * 64 + wave * 16;
  _Float16* Pw = Pb + wave * (16 * 40);

  auto stage_chunk = [&](int kc, int buf) {
    const int key = t >> 2;          // 0..31
    const int dkb = (t & 3) * 16;    // 0,16,32,48
    const size_t src = ((size_t)b * SS + kc + key) * DD + h * DKK + dkb;
    stage16(&K[src],     &Ks[buf][key * 72 + dkb]);
    stage16(&K[src] + 8, &Ks[buf][key * 72 + dkb + 8]);
    const _Float16* vp = &V[src];
    v8h v0 = *(const v8h*)vp;
    v8h v1 = *(const v8h*)(vp + 8);
#pragma unroll
    for (int i = 0; i < 8; ++i) {
      Vt[buf][(dkb + i) * 40 + key]     = v0[i];
      Vt[buf][(dkb + 8 + i) * 40 + key] = v1[i];
    }
  };

  // Q fragments (pre-scaled by 1/sqrt(DK) = 0.125, exact in f16)
  const _Float16 qs = (_Float16)0.125f;
  const _Float16* qp0 = Q + ((size_t)b * SS + (q0 + lh)) * DD + h * DKK;
  v16h aq[2];
#pragma unroll
  for (int f = 0; f < 2; ++f) {
    const _Float16* p = qp0 + f * 32 + half * 8;
    v8h lo = *(const v8h*)p;
    v8h hi = *(const v8h*)(p + 16);
#pragma unroll
    for (int i = 0; i < 8; ++i) { aq[f][i] = lo[i] * qs; aq[f][8 + i] = hi[i] * qs; }
  }

  float mrow[8], lrow[8];
  v8f oacc[4];
#pragma unroll
  for (int r = 0; r < 8; ++r) { mrow[r] = -INFINITY; lrow[r] = 0.f; }
#pragma unroll
  for (int n = 0; n < 4; ++n)
#pragma unroll
    for (int i = 0; i < 8; ++i) oacc[n][i] = 0.f;

  stage_chunk(0, 0);  // prologue

  for (int kc = 0; kc < SS; kc += 32) {
    const int cur = (kc >> 5) & 1;
    stage_wait();
    __syncthreads();
    if (kc + 32 < SS) stage_chunk(kc + 32, cur ^ 1);  // overlap next chunk

    // scores: S[16 x 32] = Qw(16x64) @ K^T(64x32), two 16x16 C tiles
    v8f s0, s1;
#pragma unroll
    for (int i = 0; i < 8; ++i) { s0[i] = 0.f; s1[i] = 0.f; }
#pragma unroll
    for (int f = 0; f < 2; ++f)
#pragma unroll
      for (int nt = 0; nt < 2; ++nt) {
        const _Float16* kp = &Ks[cur][(nt * 16 + lh) * 72 + f * 32 + half * 8];
        v8h lo = *(const v8h*)kp;
        v8h hi = *(const v8h*)(kp + 16);
        v16h bk;
#pragma unroll
        for (int i = 0; i < 8; ++i) { bk[i] = lo[i]; bk[8 + i] = hi[i]; }
        if (nt == 0)
          s0 = __builtin_amdgcn_wmma_f32_16x16x32_f16(false, aq[f], false, bk,
                                                      (short)0, s0, false, false);
        else
          s1 = __builtin_amdgcn_wmma_f32_16x16x32_f16(false, aq[f], false, bk,
                                                      (short)0, s1, false, false);
      }

    // packed mask: one u32 covers this row's 32 keys
#pragma unroll
    for (int r = 0; r < 8; ++r) {
      const size_t q = (size_t)(q0 + r + 8 * half);
      const unsigned mw = Mpk[((size_t)b * SS + q) * (SS / 32) + (kc >> 5)];
      if (!((mw >> lh) & 1u))        s0[r] = -1e9f;
      if (!((mw >> (16 + lh)) & 1u)) s1[r] = -1e9f;
    }

    // online softmax (row stats replicated across each 16-lane half-group)
#pragma unroll
    for (int r = 0; r < 8; ++r) {
      float cm = fmaxf(s0[r], s1[r]);
      for (int off = 8; off >= 1; off >>= 1)
        cm = fmaxf(cm, __shfl_xor(cm, off, 16));
      const float mnew = fmaxf(mrow[r], cm);
      const float corr = __expf(mrow[r] - mnew);
      mrow[r] = mnew;
      const float p0 = __expf(s0[r] - mnew);
      const float p1 = __expf(s1[r] - mnew);
      float rs = p0 + p1;
      for (int off = 8; off >= 1; off >>= 1) rs += __shfl_xor(rs, off, 16);
      lrow[r] = lrow[r] * corr + rs;
#pragma unroll
      for (int n = 0; n < 4; ++n) oacc[n][r] *= corr;
      const int prow = r + 8 * half;  // C-layout -> LDS bounce for A-layout
      Pw[prow * 40 + lh]      = (_Float16)p0;
      Pw[prow * 40 + 16 + lh] = (_Float16)p1;
    }

    // reload P as a 16x32 A fragment (same-wave LDS, no barrier needed)
    v16h ap;
    {
      const _Float16* pp = Pw + lh * 40 + half * 8;
      v8h lo = *(const v8h*)pp;
      v8h hi = *(const v8h*)(pp + 16);
#pragma unroll
      for (int i = 0; i < 8; ++i) { ap[i] = lo[i]; ap[8 + i] = hi[i]; }
    }

    // O += P @ Vchunk  (four 16x16 output tiles over dk)
#pragma unroll
    for (int nt = 0; nt < 4; ++nt) {
      const _Float16* vp = &Vt[cur][(nt * 16 + lh) * 40 + half * 8];
      v8h lo = *(const v8h*)vp;
      v8h hi = *(const v8h*)(vp + 16);
      v16h bv;
#pragma unroll
      for (int i = 0; i < 8; ++i) { bv[i] = lo[i]; bv[8 + i] = hi[i]; }
      oacc[nt] = __builtin_amdgcn_wmma_f32_16x16x32_f16(
          false, ap, false, bv, (short)0, oacc[nt], false, false);
    }
  }

  // normalize and write context (head-concat layout [b, s, h*DK + dk])
#pragma unroll
  for (int r = 0; r < 8; ++r) {
    const float inv = 1.0f / lrow[r];
    const size_t q = (size_t)(q0 + r + 8 * half);
    _Float16* op = O + ((size_t)b * SS + q) * DD + h * DKK;
#pragma unroll
    for (int nt = 0; nt < 4; ++nt)
      op[nt * 16 + lh] = (_Float16)(oacc[nt][r] * inv);
  }
}

// ---------------------------------------------------------------------------
// Host launcher
// ---------------------------------------------------------------------------
extern "C" void kernel_launch(void* const* d_in, const int* in_sizes, int n_in,
                              void* d_out, int out_size, void* d_ws,
                              size_t ws_size, hipStream_t stream) {
  const float* emb = (const float*)d_in[0];
  const unsigned char* mask = (const unsigned char*)d_in[1];
  const float* Wq = (const float*)d_in[2];
  const float* Wk = (const float*)d_in[3];
  const float* Wv = (const float*)d_in[4];
  const float* Wo = (const float*)d_in[5];
  float* out = (float*)d_out;

  const int M = BB * SS;  // 8192
  const int N = DD;       // 1024
  const int Kd = DD;      // 1024
  const size_t MiB = 1u << 20;

  char* ws = (char*)d_ws;
  _Float16* Xh   = (_Float16*)(ws + 0 * MiB);   // 16 MiB
  _Float16* Wqh  = (_Float16*)(ws + 16 * MiB);  //  2 MiB
  _Float16* Wkh  = (_Float16*)(ws + 18 * MiB);
  _Float16* Wvh  = (_Float16*)(ws + 20 * MiB);
  _Float16* Woh  = (_Float16*)(ws + 22 * MiB);
  _Float16* Qh   = (_Float16*)(ws + 24 * MiB);  // 16 MiB
  _Float16* Kh   = (_Float16*)(ws + 40 * MiB);  // 16 MiB
  _Float16* Vh   = (_Float16*)(ws + 56 * MiB);  // 16 MiB
  _Float16* Ctxh = (_Float16*)(ws + 72 * MiB);  // 16 MiB
  unsigned*  Mpk = (unsigned*)(ws + 88 * MiB);  //  2 MiB (total 90 MiB)

  const size_t nX = (size_t)M * Kd;
  const size_t nW = (size_t)DD * DD;
  f32_to_f16_kernel<<<512, 256, 0, stream>>>(emb, Xh, nX);
  f32_to_f16_kernel<<<256, 256, 0, stream>>>(Wq, Wqh, nW);
  f32_to_f16_kernel<<<256, 256, 0, stream>>>(Wk, Wkh, nW);
  f32_to_f16_kernel<<<256, 256, 0, stream>>>(Wv, Wvh, nW);
  f32_to_f16_kernel<<<256, 256, 0, stream>>>(Wo, Woh, nW);
  mask_pack_kernel<<<1024, 256, 0, stream>>>(mask, Mpk,
                                             (size_t)BB * SS * SS);

  dim3 gg(N / 128, M / 256);  // (8, 32)
  gemm_xwT_kernel<_Float16><<<gg, 256, 0, stream>>>(Xh, Wqh, Qh, M, N, Kd);
  gemm_xwT_kernel<_Float16><<<gg, 256, 0, stream>>>(Xh, Wkh, Kh, M, N, Kd);
  gemm_xwT_kernel<_Float16><<<gg, 256, 0, stream>>>(Xh, Wvh, Vh, M, N, Kd);

  flash_attn_kernel<<<dim3(SS / 64, HH, BB), 128, 0, stream>>>(Qh, Kh, Vh,
                                                               Mpk, Ctxh);

  gemm_xwT_kernel<float><<<gg, 256, 0, stream>>>(Ctxh, Woh, out, M, N, Kd);
}
